// CrossModalAttention2d_38989713113204
// MI455X (gfx1250) — matrix operand order repaired and checked by reading, not compiled, not run on hardware
//
#include <hip/hip_runtime.h>

// ---------------------------------------------------------------------------
// CrossModalAttention2d for MI455X (gfx1250), bf16 WMMA pipeline.
// B=32, C=512, Nq=Nk=1024, Cqk=64.
// Workspace layout (bytes):
//   [0,        4 MiB)  Qbf : bf16 [B][1024][64]  (q[b][n][cqk], pre-scaled 1/8)
//   [4 MiB,    8 MiB)  Kbf : bf16 [B][1024][64]  (k^T: [m][cqk])
//   [8 MiB,   40 MiB)  Vp  : bf16 pair-interleaved v^T:
//                            elem (m,c) at b*512K + (m>>1)*1024 + 2c + (m&1)
// CDNA5 paths: v_wmma_f32_16x16x32_bf16 for all GEMMs, fused DPP16 ALU
// butterfly reductions (v_max_num_f32_dpp / v_add_f32_dpp) for softmax,
// double-buffered global_load_async_to_lds_b128 (ASYNCcnt) pipelined
// against WMMA.
// ---------------------------------------------------------------------------

typedef __attribute__((ext_vector_type(16))) __bf16 v16bf;
typedef __attribute__((ext_vector_type(8)))  float  v8f;

union BFrag16 { v16bf v; unsigned short u[16]; unsigned int w[8]; };
union F8      { v8f   v; float f[8]; };

__device__ __forceinline__ unsigned short f2bf(float x) {
  unsigned int u = __float_as_uint(x);
  unsigned int r = (u + 0x7FFFu + ((u >> 16) & 1u)) >> 16;  // RNE
  return (unsigned short)r;
}

// ---- fused DPP16 ALU all-reduce within 16-lane row groups -----------------
// One VALU instruction per butterfly stage: op(dpp(src0), src1).
#define DPP_STAGE(OPC, PERM, x)                                               \
  {                                                                           \
    float t_;                                                                 \
    asm(OPC " %0, %1, %2 " PERM " row_mask:0xf bank_mask:0xf bound_ctrl:1"    \
        : "=v"(t_) : "v"(x), "v"(x));                                         \
    x = t_;                                                                   \
  }

__device__ __forceinline__ float row16_max(float x) {
  DPP_STAGE("v_max_num_f32_dpp", "quad_perm:[1,0,3,2]", x)  // xor 1
  DPP_STAGE("v_max_num_f32_dpp", "quad_perm:[2,3,0,1]", x)  // xor 2
  DPP_STAGE("v_max_num_f32_dpp", "row_half_mirror", x)      // xor 7 in 8
  DPP_STAGE("v_max_num_f32_dpp", "row_mirror", x)           // xor 15 in 16
  return x;
}
__device__ __forceinline__ float row16_sum(float x) {
  DPP_STAGE("v_add_f32_dpp", "quad_perm:[1,0,3,2]", x)
  DPP_STAGE("v_add_f32_dpp", "quad_perm:[2,3,0,1]", x)
  DPP_STAGE("v_add_f32_dpp", "row_half_mirror", x)
  DPP_STAGE("v_add_f32_dpp", "row_mirror", x)
  return x;
}

// ---- gfx1250 async global -> LDS copy (16B per lane, ASYNCcnt tracked) ----
__device__ __forceinline__ void async_copy_b128(unsigned lds_off,
                                                const void* gaddr) {
  asm volatile("global_load_async_to_lds_b128 %0, %1, off"
               :: "v"(lds_off), "v"(gaddr) : "memory");
}
template <int NLeft>
__device__ __forceinline__ void wait_async() {
  asm volatile("s_wait_asynccnt %0" :: "i"(NLeft) : "memory");
}

static constexpr int Bsz = 32, C = 512, N = 1024, CQK = 64;

// ---------------------------------------------------------------------------
// Projection GEMM: Out[b][n][j] = bf16( (sum_c In[b][c][n]*W[j][c] + b[j])*s )
// WG = 256 threads = 8 waves; WG tile 64(n) x 64(j); wave tile 16(n) x 32(j).
// PACK=true stores pair-interleaved rows (for the attention V operand).
// ---------------------------------------------------------------------------
template <int J, bool PACK>
__global__ __launch_bounds__(256) void proj_gemm_kernel(
    const float* __restrict__ In, const float* __restrict__ W,
    const float* __restrict__ bias, unsigned short* __restrict__ Out,
    float outScale) {
  __shared__ __align__(16) unsigned short Atile[64 * 32];  // [n][c] bf16

  const int tid  = threadIdx.x;
  const int wave = tid >> 5, lane = tid & 31;
  const int half = lane >> 4, l15 = lane & 15;
  const int wn = wave & 3, wj = wave >> 2;

  const int b  = blockIdx.x >> 4;          // 16 n-tiles per batch
  const int n0 = (blockIdx.x & 15) * 64;
  const int j0 = blockIdx.y * 64;

  F8 acc[2];
  acc[0].v = (v8f){};
  acc[1].v = (v8f){};

  const int ci = tid >> 3;                 // stage: 32 c-rows, 8 threads/row
  const int nb = (tid & 7) * 8;

  for (int cbase = 0; cbase < C; cbase += 32) {
    // ---- stage A: In[b][cbase+ci][n0+nn] -> Atile[nn][ci] (bf16, transposed)
    const float* src = In + ((size_t)b * C + (cbase + ci)) * N + n0 + nb;
#pragma unroll
    for (int i = 0; i < 8; ++i)
      Atile[(nb + i) * 32 + ci] = f2bf(src[i]);
    __syncthreads();

    // ---- A fragment (16x32 bf16): M = l15, contiguous K = 16*half + s
    BFrag16 afr;
    const int arow = (wn * 16 + l15) * 32;
#pragma unroll
    for (int v = 0; v < 8; ++v) {
      const int kk = half * 16 + 2 * v;
      afr.w[v] = *(const unsigned int*)&Atile[arow + kk];
    }

    // ---- B fragments (32x16): B[kk][j] = W[j][cbase+kk], f32->bf16 on the fly
#pragma unroll
    for (int jsub = 0; jsub < 2; ++jsub) {
      const int j = j0 + wj * 32 + jsub * 16 + l15;
      BFrag16 bfr;
#pragma unroll
      for (int v = 0; v < 8; ++v) {
        const int cc = cbase + half * 16 + 2 * v;
        const float* wp = W + (size_t)j * C + cc;
        bfr.w[v] = (unsigned int)f2bf(wp[0]) | ((unsigned int)f2bf(wp[1]) << 16);
      }
      acc[jsub].v = __builtin_amdgcn_wmma_f32_16x16x32_bf16(
          false, afr.v, false, bfr.v, (short)0, acc[jsub].v, false, false);
    }
    __syncthreads();
  }

  // ---- epilogue: (+bias)*scale, store bf16
#pragma unroll
  for (int jsub = 0; jsub < 2; ++jsub) {
    const int j = j0 + wj * 32 + jsub * 16 + l15;
    const float bj = bias[j];
#pragma unroll
    for (int v = 0; v < 8; ++v) {
      const int n = n0 + wn * 16 + v + 8 * half;
      const size_t off = PACK
          ? (size_t)b * N * J + (size_t)(n >> 1) * (2 * J) + 2 * j + (n & 1)
          : ((size_t)b * N + n) * J + j;
      Out[off] = f2bf((acc[jsub].f[v] + bj) * outScale);
    }
  }
}

// ---------------------------------------------------------------------------
// Fused flash attention: out[b][c][n] = gamma * (softmax(QK^T) V^T)[n][c]
//                                       + face[b][c][n]   (Q pre-scaled 1/8)
// WG tile: 32(q) x 128(c); 8 waves = 2(q) x 4(c), wave tile 16(q) x 32(c).
// Double-buffered async LDS staging pipelined against WMMA.
// ---------------------------------------------------------------------------
__global__ __launch_bounds__(256) void attn_kernel(
    const unsigned short* __restrict__ Qbf, const unsigned short* __restrict__ Kbf,
    const unsigned short* __restrict__ Vp, const float* __restrict__ face,
    const float* __restrict__ gamma, float* __restrict__ out) {
  __shared__ __align__(16) unsigned short Ktile[2][32 * 64];    // [m][cqk]
  __shared__ __align__(16) unsigned short Vtile[2][16 * 256];   // [m>>1][2c|m&1]
  __shared__ __align__(16) unsigned short Ptile[8 * 16 * 32];   // per-wave [q][m]

  const int tid  = threadIdx.x;
  const int wave = tid >> 5, lane = tid & 31;
  const int half = lane >> 4, l15 = lane & 15;
  const int wq = wave & 1, wc = wave >> 1;

  const int b  = blockIdx.x >> 5;                  // 32 q-tiles per batch
  const int n0 = (blockIdx.x & 31) * 32;
  const int c0 = blockIdx.y * 128;

  // ---- Q fragments (two K-chunks of 32 over cqk=64), loaded once
  BFrag16 qa[2];
  {
    const int nrow = n0 + wq * 16 + l15;
    const unsigned short* qp = Qbf + ((size_t)b * N + nrow) * CQK;
#pragma unroll
    for (int kc = 0; kc < 2; ++kc)
#pragma unroll
      for (int v = 0; v < 8; ++v) {
        const int kk = kc * 32 + half * 16 + 2 * v;
        qa[kc].w[v] = *(const unsigned int*)(qp + kk);
      }
  }

  // per-lane async-copy coordinates (16B chunks)
  const int kmm = tid >> 3, kco = (tid & 7) * 8;   // K: 32 rows x 8 chunks
  const int vpr = tid >> 4, vco = (tid & 15) * 16; // V: 16 prows x 16 chunks
  const unsigned klds[2] = {(unsigned)(size_t)&Ktile[0][kmm * 64 + kco],
                            (unsigned)(size_t)&Ktile[1][kmm * 64 + kco]};
  const unsigned vlds[2] = {(unsigned)(size_t)&Vtile[0][vpr * 256 + vco],
                            (unsigned)(size_t)&Vtile[1][vpr * 256 + vco]};
  const unsigned short* kgb = Kbf + ((size_t)b * N + kmm) * CQK + kco;
  const unsigned short* vgb = Vp + (size_t)b * N * C + (size_t)vpr * 1024
                                 + c0 * 2 + vco;

  float Mrow[8], Lrow[8];
  F8 accO[2];
  accO[0].v = (v8f){};
  accO[1].v = (v8f){};
#pragma unroll
  for (int v = 0; v < 8; ++v) { Mrow[v] = -1e30f; Lrow[v] = 0.0f; }

  unsigned short* Pw = Ptile + wave * (16 * 32);

  // ---- prologue: stage tile 0 into buffer 0 (3 async ops / thread)
  async_copy_b128(klds[0], kgb);
  async_copy_b128(vlds[0], vgb);
  async_copy_b128(vlds[0] + 16, vgb + 8);

  for (int it = 0; it < N / 32; ++it) {
    const int cur = it & 1;
    if (it != N / 32 - 1) {
      // issue next tile into the other buffer; overlap with this tile's math
      const int nxt = 1 - cur;
      async_copy_b128(klds[nxt], kgb + (size_t)(it + 1) * 32 * CQK);
      async_copy_b128(vlds[nxt], vgb + (size_t)(it + 1) * 16 * 1024);
      async_copy_b128(vlds[nxt] + 16, vgb + (size_t)(it + 1) * 16 * 1024 + 8);
      wait_async<3>();   // previous 3 (this tile) complete; next 3 in flight
    } else {
      wait_async<0>();
    }
    __syncthreads();

    const unsigned short* Kt = Ktile[cur];
    const unsigned short* Vt = Vtile[cur];

    // ---- S = Q K^T (pre-scaled), two 16x16 m-subtiles
    F8 S[2];
#pragma unroll
    for (int msub = 0; msub < 2; ++msub) {
      S[msub].v = (v8f){};
#pragma unroll
      for (int kc = 0; kc < 2; ++kc) {
        BFrag16 bk;
        const int mcol = msub * 16 + l15;
#pragma unroll
        for (int v = 0; v < 8; ++v) {
          const int kk = kc * 32 + half * 16 + 2 * v;
          bk.w[v] = *(const unsigned int*)&Kt[mcol * 64 + kk];
        }
        S[msub].v = __builtin_amdgcn_wmma_f32_16x16x32_bf16(
            false, qa[kc].v, false, bk.v, (short)0, S[msub].v, false, false);
      }
    }

    // ---- online softmax per row (row = v + 8*half; cols in 16-lane group)
#pragma unroll
    for (int v = 0; v < 8; ++v) {
      const float s0 = S[0].f[v], s1 = S[1].f[v];
      const float mx = row16_max(fmaxf(s0, s1));
      const float newM = fmaxf(Mrow[v], mx);
      const float alpha = __expf(Mrow[v] - newM);
      Mrow[v] = newM;
      const float p0 = __expf(fmaxf(s0 - newM, -50.0f));
      const float p1 = __expf(fmaxf(s1 - newM, -50.0f));
      const float rs = row16_sum(p0 + p1);
      Lrow[v] = Lrow[v] * alpha + rs;
      accO[0].f[v] *= alpha;
      accO[1].f[v] *= alpha;
      const int r = v + 8 * half;  // C-layout row -> P staging row
      Pw[r * 32 + l15]      = f2bf(p0);
      Pw[r * 32 + 16 + l15] = f2bf(p1);
    }
    // per-wave LDS region; same-wave DS ops are in-order: no barrier needed

    // ---- O += P (16x32) @ V (32x128 slice); V pairs contiguous in LDS
    BFrag16 pa;
#pragma unroll
    for (int v = 0; v < 8; ++v) {
      const int kk = half * 16 + 2 * v;
      pa.w[v] = *(const unsigned int*)&Pw[l15 * 32 + kk];
    }
#pragma unroll
    for (int csub = 0; csub < 2; ++csub) {
      const int col = wc * 32 + csub * 16 + l15;
      BFrag16 bv;
#pragma unroll
      for (int v = 0; v < 8; ++v) {
        const int mpair = half * 8 + v;  // rows (2*mpair, 2*mpair+1)
        bv.w[v] = *(const unsigned int*)&Vt[mpair * 256 + col * 2];
      }
      accO[csub].v = __builtin_amdgcn_wmma_f32_16x16x32_bf16(
          false, pa.v, false, bv.v, (short)0, accO[csub].v, false, false);
    }
    __syncthreads();  // all reads of buffer `cur` done before it is refilled
  }

  // ---- epilogue: out[b][c][n] = gamma * O/L + face[b][c][n]
  const float g = gamma[0];
#pragma unroll
  for (int csub = 0; csub < 2; ++csub) {
#pragma unroll
    for (int v = 0; v < 8; ++v) {
      const int n = n0 + wq * 16 + v + 8 * half;
      const int c = c0 + wc * 32 + csub * 16 + l15;
      const size_t idx = ((size_t)b * C + c) * N + n;
      out[idx] = g * (accO[csub].f[v] / Lrow[v]) + face[idx];
    }
  }
}

// ---------------------------------------------------------------------------
extern "C" void kernel_launch(void* const* d_in, const int* in_sizes, int n_in,
                              void* d_out, int out_size, void* d_ws, size_t ws_size,
                              hipStream_t stream) {
  const float* face  = (const float*)d_in[0];
  const float* audio = (const float*)d_in[1];
  const float* Wq    = (const float*)d_in[2];
  const float* bq    = (const float*)d_in[3];
  const float* Wk    = (const float*)d_in[4];
  const float* bk    = (const float*)d_in[5];
  const float* Wv    = (const float*)d_in[6];
  const float* bv    = (const float*)d_in[7];
  const float* gamma = (const float*)d_in[8];
  float* out = (float*)d_out;

  unsigned short* Qbf = (unsigned short*)d_ws;                       // 4 MiB
  unsigned short* Kbf = (unsigned short*)((char*)d_ws + (4u << 20)); // 4 MiB
  unsigned short* Vp  = (unsigned short*)((char*)d_ws + (8u << 20)); // 32 MiB

  // Projections: Q (face, pre-scaled 1/sqrt(64)), K^T (audio), V^T packed (audio)
  proj_gemm_kernel<64, false><<<dim3(Bsz * 16, 1), 256, 0, stream>>>(face,  Wq, bq, Qbf, 0.125f);
  proj_gemm_kernel<64, false><<<dim3(Bsz * 16, 1), 256, 0, stream>>>(audio, Wk, bk, Kbf, 1.0f);
  proj_gemm_kernel<512, true><<<dim3(Bsz * 16, 8), 256, 0, stream>>>(audio, Wv, bv, Vp, 1.0f);

  // Fused flash attention + residual
  attn_kernel<<<dim3(Bsz * 32, 4), 256, 0, stream>>>(Qbf, Kbf, Vp, face, gamma, out);
}
